// Attention_37280316129462
// MI455X (gfx1250) — compile-verified, matrix operand order checked by reading, not run on hardware
//
#include <hip/hip_runtime.h>
#include <hip/hip_bf16.h>

// ---------------------------------------------------------------------------
// GPT-2 attention block on CDNA5 (gfx1250), all GEMM work on v_wmma bf16.
//   B=2, S=2048, D=1024, H=16, dk=64.  fp32 in/out, bf16 matrix operands,
//   fp32 WMMA accumulation (V_WMMA_F32_16X16X32_BF16).
// ---------------------------------------------------------------------------

typedef __bf16 bf16;
typedef __attribute__((ext_vector_type(16))) __bf16 v16bf;
typedef __attribute__((ext_vector_type(8)))  __bf16 v8bf;
typedef __attribute__((ext_vector_type(4)))  __bf16 v4bf;
typedef __attribute__((ext_vector_type(8)))  float  v8f;

#define B_   2
#define S_   2048
#define D_   1024
#define H_   16
#define DK_  64
#define BS_  (B_ * S_)       // 4096 rows
#define D3_  (3 * D_)        // 3072

static constexpr float NEG_BIG = -3.0e38f;

// 16-bit WMMA A/B fragment (ISA 7.12.2): lane holds row/col = lane%16,
// K-chunks [kb, kb+8) and [kb+16, kb+24) where kb = (lane>=16)?8:0 within the
// 32-wide contraction window.  Both chunks are 16B-contiguous in K-major data.
__device__ __forceinline__ v16bf load_frag(const bf16* p) {
  v8bf lo = *(const v8bf*)p;
  v8bf hi = *(const v8bf*)(p + 16);
  return __builtin_shufflevector(lo, hi, 0,1,2,3,4,5,6,7,8,9,10,11,12,13,14,15);
}

__device__ __forceinline__ v8f wmma_bf16(v16bf a, v16bf b, v8f c) {
  return __builtin_amdgcn_wmma_f32_16x16x32_bf16(
      /*neg_a=*/false, a, /*neg_b=*/false, b,
      /*c_mod=*/(short)0, c, /*reuse_a=*/false, /*reuse_b=*/false);
}

// ---------------------------------------------------------------------------
// fp32 -> bf16 elementwise (vectorized x4)
// ---------------------------------------------------------------------------
__global__ __launch_bounds__(256)
void convert_bf16_kernel(const float* __restrict__ in, bf16* __restrict__ out,
                         int n4) {
  int i = blockIdx.x * 256 + threadIdx.x;
  if (i < n4) {
    float4 v = ((const float4*)in)[i];
    v4bf o;
    o[0] = (bf16)v.x; o[1] = (bf16)v.y; o[2] = (bf16)v.z; o[3] = (bf16)v.w;
    ((v4bf*)out)[i] = o;
  }
}

// ---------------------------------------------------------------------------
// fp32 [R,C] -> bf16 [C,R]  (transpose so GEMM B-operand is K-major)
// ---------------------------------------------------------------------------
__global__ __launch_bounds__(256)
void convert_transpose_kernel(const float* __restrict__ in,
                              bf16* __restrict__ out, int R, int C) {
  __shared__ bf16 tile[32][33];
  int cBase = blockIdx.x * 32, rBase = blockIdx.y * 32;
  int tx = threadIdx.x & 31, ty = threadIdx.x >> 5;   // 8 rows per pass
#pragma unroll
  for (int i = 0; i < 32; i += 8)
    tile[ty + i][tx] = (bf16)in[(size_t)(rBase + ty + i) * C + cBase + tx];
  __syncthreads();
#pragma unroll
  for (int i = 0; i < 32; i += 8)
    out[(size_t)(cBase + ty + i) * R + rBase + tx] = tile[tx][ty + i];
}

// ---------------------------------------------------------------------------
// C[M,N] = A[M,K] @ Bt[N,K]^T + bias[N]
// 256 threads = 8 wave32s; block tile 128x128; K-step 64 staged in LDS
// (stride 72 bf16 = 144 B = 36 banks; 36*r mod 64 distinct & =0 mod 4 for
// r=0..15 -> 16-B fragment loads are bank-conflict free).
// Each wave: 32x64 output = 2x4 tiles; 16 v_wmma per barrier pair.
// Next K-tile is prefetched (global_prefetch_b8) during compute.
// ---------------------------------------------------------------------------
template <bool OUT_BF16>
__global__ __launch_bounds__(256)
void gemm_nt_bias_kernel(const bf16* __restrict__ A, const bf16* __restrict__ Bt,
                         const float* __restrict__ bias, void* __restrict__ Cout,
                         int M, int N, int K) {
  constexpr int LDT = 72;
  __shared__ bf16 As[128 * LDT];
  __shared__ bf16 Bs[128 * LDT];

  const int tid = threadIdx.x;
  const int wave = tid >> 5, lane = tid & 31;
  const int laneMod = lane & 15, laneHi = lane >> 4;
  const int blockM = blockIdx.y * 128, blockN = blockIdx.x * 128;
  const int waveM = (wave >> 1) * 32;   // 4 waves across M
  const int waveN = (wave & 1) * 64;    // 2 waves across N

  v8f acc[2][4] = {};

  for (int k0 = 0; k0 < K; k0 += 64) {
    // Cooperative stage: 1024 chunks of 8 bf16 per matrix, 4 chunks/thread.
#pragma unroll
    for (int c = tid; c < 1024; c += 256) {
      int r = c >> 3, q = (c & 7) * 8;
      *(v8bf*)&As[r * LDT + q] =
          *(const v8bf*)&A[(size_t)(blockM + r) * K + k0 + q];
      *(v8bf*)&Bs[r * LDT + q] =
          *(const v8bf*)&Bt[(size_t)(blockN + r) * K + k0 + q];
      if (k0 + 64 < K) {   // hide next tile's L2 latency
        __builtin_prefetch(&A[(size_t)(blockM + r) * K + k0 + 64 + q], 0, 1);
        __builtin_prefetch(&Bt[(size_t)(blockN + r) * K + k0 + 64 + q], 0, 1);
      }
    }
    __syncthreads();

    const int kb = laneHi * 8;
#pragma unroll
    for (int s = 0; s < 2; ++s) {       // two 32-wide contraction windows
      v16bf af[2], bfg[4];
#pragma unroll
      for (int mi = 0; mi < 2; ++mi)
        af[mi] = load_frag(&As[(waveM + mi * 16 + laneMod) * LDT + s * 32 + kb]);
#pragma unroll
      for (int ni = 0; ni < 4; ++ni)
        bfg[ni] = load_frag(&Bs[(waveN + ni * 16 + laneMod) * LDT + s * 32 + kb]);
#pragma unroll
      for (int mi = 0; mi < 2; ++mi)
#pragma unroll
        for (int ni = 0; ni < 4; ++ni)
          acc[mi][ni] = wmma_bf16(af[mi], bfg[ni], acc[mi][ni]);
    }
    __syncthreads();
  }

  // Epilogue: C layout -> lane = column (lane%16), VGPR i = row i + 8*(lane/16)
#pragma unroll
  for (int mi = 0; mi < 2; ++mi)
#pragma unroll
    for (int ni = 0; ni < 4; ++ni) {
      int col = blockN + waveN + ni * 16 + laneMod;
      float bv = bias[col];
#pragma unroll
      for (int i = 0; i < 8; ++i) {
        int row = blockM + waveM + mi * 16 + i + 8 * laneHi;
        float v = acc[mi][ni][i] + bv;
        if (OUT_BF16)
          ((bf16*)Cout)[(size_t)row * N + col] = (bf16)v;
        else
          ((float*)Cout)[(size_t)row * N + col] = v;
      }
    }
}

// ---------------------------------------------------------------------------
// Causal flash attention.  qkv: bf16 [B*S, 3D] (q|k|v), out: bf16 [B*S, D].
// Block = 256 threads (8 waves) handles (b, h, 128 query rows); each wave owns
// 16 query rows.  KV tiles of 32 staged in LDS (V transposed to dk-major).
// QK^T: batched fragment loads then 4 WMMA; fully unrolled online softmax
// (row stats reduce over 16-lane groups via __shfl_xor, 8 independent
// chains interleave); P re-shaped through wave-private LDS; P@V: 4 WMMA.
// Output bounced through LDS for coalesced 16-B stores.
// ---------------------------------------------------------------------------
__global__ __launch_bounds__(256)
void flash_attn_kernel(const bf16* __restrict__ qkv, bf16* __restrict__ out) {
  const int bh = blockIdx.x;
  const int b = bh / H_, h = bh % H_;
  const int qBase = blockIdx.y * 128;
  const int tid = threadIdx.x, wave = tid >> 5, lane = tid & 31;
  const int laneMod = lane & 15, laneHi = lane >> 4;

  __shared__ bf16 Ks[32 * 72];        // [key][dk]   stride 72 (144 B)
  __shared__ bf16 Vt[64 * 40];        // [dk][key]   transposed, stride 40
  __shared__ bf16 Ps[8][16 * 72];     // per-wave: P staging (stride 40) and
                                      // final O staging (stride 72)

  const size_t rowStride = (size_t)D3_;
  const size_t baseRow = (size_t)b * S_;

  // Q fragments for this wave's 16 rows, both dk windows (0..31, 32..63).
  const int qRow = qBase + wave * 16 + laneMod;
  v16bf qf[2];
#pragma unroll
  for (int s = 0; s < 2; ++s) {
    int kb = s * 32 + laneHi * 8;
    qf[s] = load_frag(&qkv[(baseRow + qRow) * rowStride + h * DK_ + kb]);
  }

  float m_i[8], l_i[8];
  v8f o[4] = {};
#pragma unroll
  for (int i = 0; i < 8; ++i) { m_i[i] = NEG_BIG; l_i[i] = 0.f; }

  const int nTiles = (qBase + 128) / 32;   // causal: kv <= last q row of block
  const float scale = 0.125f;              // 1/sqrt(64)
  const int ldr = tid >> 3, ldq = (tid & 7) * 8;   // 32 keys x 8 dk-chunks

  for (int t = 0; t < nTiles; ++t) {
    const int kvBase = t * 32;
    // Stage K tile (straight copy) and V tile (transpose scatter).
    {
      size_t row = (baseRow + kvBase + ldr) * rowStride + h * DK_;
      *(v8bf*)&Ks[ldr * 72 + ldq] = *(const v8bf*)&qkv[row + D_ + ldq];
      v8bf vv = *(const v8bf*)&qkv[row + 2 * D_ + ldq];
#pragma unroll
      for (int j = 0; j < 8; ++j) Vt[(ldq + j) * 40 + ldr] = vv[j];
      if (t + 1 < nTiles) {            // prefetch next KV tile into L2/L0
        size_t nrow = (baseRow + kvBase + 32 + ldr) * rowStride + h * DK_;
        __builtin_prefetch(&qkv[nrow + D_ + ldq], 0, 1);
        __builtin_prefetch(&qkv[nrow + 2 * D_ + ldq], 0, 1);
      }
    }
    __syncthreads();

    // Scores S = Q K^T (16 q x 32 kv): batch all 4 fragment loads, then WMMA.
    v8f s0 = {}, s1 = {};
    {
      const int kb = laneHi * 8;
      v16bf kf[2][2];
#pragma unroll
      for (int s = 0; s < 2; ++s) {
        kf[s][0] = load_frag(&Ks[(laneMod)      * 72 + s * 32 + kb]);
        kf[s][1] = load_frag(&Ks[(16 + laneMod) * 72 + s * 32 + kb]);
      }
#pragma unroll
      for (int s = 0; s < 2; ++s) {
        s0 = wmma_bf16(qf[s], kf[s][0], s0);
        s1 = wmma_bf16(qf[s], kf[s][1], s1);
      }
    }

    // Causal mask + online softmax.  C layout: lane holds column laneMod,
    // VGPR i holds row i + 8*laneHi.  Fully unrolled: 8 independent
    // shuffle-reduction chains overlap.
    const int c0 = kvBase + laneMod, c1 = kvBase + 16 + laneMod;
    const int rBase = qBase + wave * 16 + 8 * laneHi;
    float rowScale[8];
#pragma unroll
    for (int i = 0; i < 8; ++i) {
      int r = rBase + i;
      float a = s0[i] * scale, bb = s1[i] * scale;
      if (c0 > r) a  = NEG_BIG;
      if (c1 > r) bb = NEG_BIG;
      float mx = fmaxf(a, bb);
#pragma unroll
      for (int off = 8; off; off >>= 1) mx = fmaxf(mx, __shfl_xor(mx, off, 32));
      float mNew = fmaxf(m_i[i], mx);
      float pa = __expf(a - mNew), pb = __expf(bb - mNew);
      float rs = pa + pb;
#pragma unroll
      for (int off = 8; off; off >>= 1) rs += __shfl_xor(rs, off, 32);
      float corr = __expf(m_i[i] - mNew);
      l_i[i] = l_i[i] * corr + rs;
      m_i[i] = mNew;
      rowScale[i] = corr;
      // Reshape P (C layout) -> A-fragment layout via wave-private LDS.
      Ps[wave][(i + 8 * laneHi) * 40 + laneMod]      = (bf16)pa;
      Ps[wave][(i + 8 * laneHi) * 40 + 16 + laneMod] = (bf16)pb;
    }
#pragma unroll
    for (int n = 0; n < 4; ++n)
#pragma unroll
      for (int i = 0; i < 8; ++i) o[n][i] *= rowScale[i];

    // O += P @ V (contraction over 32 keys; DS ops in-order within wave).
    {
      const int kvb = laneHi * 8;
      v16bf pf = load_frag(&Ps[wave][laneMod * 40 + kvb]);
      v16bf vf[4];
#pragma unroll
      for (int n = 0; n < 4; ++n)
        vf[n] = load_frag(&Vt[(n * 16 + laneMod) * 40 + kvb]);
#pragma unroll
      for (int n = 0; n < 4; ++n) o[n] = wmma_bf16(pf, vf[n], o[n]);
    }
    __syncthreads();   // protect Ks/Vt before next tile's stage
  }

  // Normalize; bounce O through wave-private LDS (stride 72) so the global
  // write is 4x 16-B coalesced stores per lane instead of 32x 2-B scatter.
#pragma unroll
  for (int i = 0; i < 8; ++i) {
    float inv = 1.0f / l_i[i];
#pragma unroll
    for (int n = 0; n < 4; ++n)
      Ps[wave][(i + 8 * laneHi) * 72 + n * 16 + laneMod] = (bf16)(o[n][i] * inv);
  }
#pragma unroll
  for (int pass = 0; pass < 4; ++pass) {
    int idx = pass * 32 + lane;        // 128 chunks: 16 rows x 8 chunks
    int rr = idx >> 3, cc = (idx & 7) * 8;
    v8bf vv = *(const v8bf*)&Ps[wave][rr * 72 + cc];
    *(v8bf*)&out[(baseRow + qBase + wave * 16 + rr) * (size_t)D_ + h * DK_ + cc] = vv;
  }
}

// ---------------------------------------------------------------------------
// Host-side orchestration (graph-capture safe: only kernel launches on stream)
// ---------------------------------------------------------------------------
extern "C" void kernel_launch(void* const* d_in, const int* in_sizes, int n_in,
                              void* d_out, int out_size, void* d_ws, size_t ws_size,
                              hipStream_t stream) {
  (void)in_sizes; (void)n_in; (void)out_size; (void)ws_size;
  const float* hidden  = (const float*)d_in[0];   // [B,S,D] f32
  const float* attn_w  = (const float*)d_in[1];   // [D,3D] f32
  const float* attn_b  = (const float*)d_in[2];   // [3D]   f32
  const float* proj_w  = (const float*)d_in[3];   // [D,D]  f32
  const float* proj_b  = (const float*)d_in[4];   // [D]    f32
  float* outp = (float*)d_out;                    // [B,S,D] f32

  // Workspace layout (bf16): 48 MB total.
  bf16* Xb    = (bf16*)d_ws;                      // [4096,1024]
  bf16* WqkvT = Xb    + (size_t)BS_ * D_;         // [3072,1024]
  bf16* WproT = WqkvT + (size_t)D3_ * D_;         // [1024,1024]
  bf16* QKV   = WproT + (size_t)D_  * D_;         // [4096,3072]
  bf16* Aout  = QKV   + (size_t)BS_ * D3_;        // [4096,1024]

  // 1) Precision conversion / weight transposition.
  {
    int n4 = (BS_ * D_) / 4;
    convert_bf16_kernel<<<(n4 + 255) / 256, 256, 0, stream>>>(hidden, Xb, n4);
    convert_transpose_kernel<<<dim3(D3_ / 32, D_ / 32), 256, 0, stream>>>(
        attn_w, WqkvT, D_, D3_);
    convert_transpose_kernel<<<dim3(D_ / 32, D_ / 32), 256, 0, stream>>>(
        proj_w, WproT, D_, D_);
  }

  // 2) QKV projection: [4096,1024] x [1024,3072] + b -> bf16 [4096,3072].
  gemm_nt_bias_kernel<true><<<dim3(D3_ / 128, BS_ / 128), 256, 0, stream>>>(
      Xb, WqkvT, attn_b, QKV, BS_, D3_, D_);

  // 3) Causal flash attention -> bf16 [4096,1024] merged heads.
  flash_attn_kernel<<<dim3(B_ * H_, S_ / 128), 256, 0, stream>>>(QKV, Aout);

  // 4) Output projection: [4096,1024] x [1024,1024] + b -> f32 d_out.
  gemm_nt_bias_kernel<false><<<dim3(D_ / 128, BS_ / 128), 256, 0, stream>>>(
      Aout, WproT, proj_b, outp, BS_, D_, D_);
}
